// LinearAttention_20160576488015
// MI455X (gfx1250) — matrix operand order repaired and checked by reading, not compiled, not run on hardware
//
#include <hip/hip_runtime.h>

// ---------------- problem constants ----------------
constexpr int Dm    = 512;          // d_model
constexpr int HEADS = 4;
constexpr int DK    = 256;          // total key dim
constexpr int DV    = 512;          // total value dim
constexpr int HK    = 64;           // per-head key dim
constexpr int HV    = 128;          // per-head value dim
constexpr int GLR   = 16;           // gate low rank
constexpr int BATCH = 2;
constexpr int SEQ   = 16384;
constexpr int NTOK  = BATCH * SEQ;  // 32768
constexpr int CHUNK = 64;
constexpr int NC    = SEQ / CHUNK;  // 256 chunks per sequence
constexpr int BH    = BATCH * HEADS;
constexpr int KP    = 72;           // padded K stride for 64-wide LDS bf16 tiles
constexpr int KP512 = 520;          // padded K stride for gemm B panel

typedef __attribute__((ext_vector_type(16))) __bf16   v16bf;
typedef __attribute__((ext_vector_type(8)))  float    v8f;
typedef __attribute__((ext_vector_type(4)))  unsigned v4u;

// ---------------- helpers ----------------
static __device__ inline __bf16 f2bf(float f) {
  unsigned u = __builtin_bit_cast(unsigned, f);
  unsigned r = u + 0x7FFFu + ((u >> 16) & 1u);   // round-to-nearest-even
  unsigned short s = (unsigned short)(r >> 16);
  return __builtin_bit_cast(__bf16, s);
}

// A fragment: base points at row start + k0 (contiguous along K). Two b128 loads.
static __device__ inline v16bf fragA(const __bf16* base, int hi) {
  union { v16bf v; v4u q[2]; } u;
  u.q[0] = *reinterpret_cast<const v4u*>(base + hi * 8);
  u.q[1] = *reinterpret_cast<const v4u*>(base + 16 + hi * 8);
  return u.v;
}
// B fragment: base points at col start + k0 in a [N][K] layout. Two b128 loads.
static __device__ inline v16bf fragB(const __bf16* base, int hi) {
  union { v16bf v; v4u q[2]; } u;
  u.q[0] = *reinterpret_cast<const v4u*>(base + hi * 8);
  u.q[1] = *reinterpret_cast<const v4u*>(base + 16 + hi * 8);
  return u.v;
}

// ---------------- f32 -> bf16 convert ----------------
__global__ void cvt_bf16_kernel(const float* __restrict__ in,
                                __bf16* __restrict__ out, size_t n) {
  size_t i = (size_t)blockIdx.x * blockDim.x + threadIdx.x;
  size_t stride = (size_t)gridDim.x * blockDim.x;
  for (; i < n; i += stride) out[i] = f2bf(in[i]);
}

// ---------------- generic bf16 WMMA GEMM: C[N,M] = alpha * A[N,K] @ B[K,M] ----------------
// block = 256 (8 waves); each wave one 16x16 tile; block covers 128 rows x 16 cols.
__global__ void gemm_bf16_kernel(const __bf16* __restrict__ A,
                                 const __bf16* __restrict__ Bm,
                                 float* __restrict__ C,
                                 int N, int K, int M, float alpha) {
  __shared__ __align__(16) __bf16 Bs[16][KP512];   // B panel, transposed [col][K]
  const int tid = threadIdx.x;
  const int col0 = blockIdx.x * 16;
  for (int e = tid; e < K * 16; e += 256) {
    int kk = e >> 4, cc = e & 15;
    Bs[cc][kk] = Bm[(size_t)kk * M + col0 + cc];
  }
  __syncthreads();

  const int w = tid >> 5, lane = tid & 31, l = lane & 15, hi = lane >> 4;
  const int row0 = blockIdx.y * 128 + w * 16;
  const __bf16* arow = A + (size_t)(row0 + l) * K;
  const __bf16* brow = &Bs[l][0];

  v8f acc = {};
  for (int k0 = 0; k0 < K; k0 += 32) {
    __builtin_prefetch(arow + k0 + 128, 0, 1);
    v16bf a = fragA(arow + k0, hi);
    v16bf b = fragB(brow + k0, hi);
    acc = __builtin_amdgcn_wmma_f32_16x16x32_bf16(false, a, false, b,
                                                  (short)0, acc, false, false);
  }
#pragma unroll
  for (int i = 0; i < 8; ++i) {
    int row = row0 + hi * 8 + i;
    C[(size_t)row * M + col0 + l] = alpha * acc[i];
  }
}

// ---------------- low-rank forget gate: gk = log_sigmoid(xg1 @ Wgk2 + b) / 16 ----------------
__global__ void gate_lowrank_kernel(const float* __restrict__ xg1,
                                    const float* __restrict__ Wgk2,
                                    const float* __restrict__ bgk2,
                                    float* __restrict__ gk) {
  size_t idx = (size_t)blockIdx.x * 256 + threadIdx.x;
  if (idx >= (size_t)NTOK * DK) return;
  int n = (int)(idx / DK), j = (int)(idx % DK);
  float s = bgk2[j];
#pragma unroll
  for (int r = 0; r < GLR; ++r) s += xg1[(size_t)n * GLR + r] * Wgk2[r * DK + j];
  float m = fminf(s, 0.f);
  float lsig = m - log1pf(__expf(-fabsf(s)));
  gk[idx] = lsig * (1.f / 16.f);
}

// ---------------- chunk-local state: S_c = Kbar^T V, d = exp(G_L), store G ----------------
// grid = (NC, BH), block = 256
__global__ void chunk_state_kernel(const float* __restrict__ kproj,
                                   const float* __restrict__ vproj,
                                   const float* __restrict__ gk,
                                   float* __restrict__ Sbuf,
                                   float* __restrict__ dvec,
                                   float* __restrict__ Gbuf) {
  const int c = blockIdx.x, bh = blockIdx.y;
  const int b = bh / HEADS, h = bh % HEADS;
  const int tok0 = b * SEQ + c * CHUNK;
  __shared__ float Gs[CHUNK][HK];                    // 16 KB cumulative log decay
  __shared__ __align__(16) __bf16 ksT[HK][KP];       // A: [i][t]
  __shared__ __align__(16) __bf16 vT[HV][KP];        // B: [j][t]
  const int tid = threadIdx.x;
  const size_t gbase = (size_t)bh * NC + c;

  for (int e = tid; e < CHUNK * HK; e += 256) {
    int t = e >> 6, i = e & 63;
    Gs[t][i] = gk[(size_t)(tok0 + t) * DK + h * HK + i];
  }
  __syncthreads();
  if (tid < HK) {                    // inclusive cumsum over time per k-dim
    float run = 0.f;
    for (int t = 0; t < CHUNK; ++t) { run += Gs[t][tid]; Gs[t][tid] = run; }
  }
  __syncthreads();
  for (int e = tid; e < CHUNK * HK; e += 256) {
    int t = e >> 6, i = e & 63;
    Gbuf[gbase * (CHUNK * HK) + e] = Gs[t][i];
  }
  if (tid < HK) dvec[gbase * HK + tid] = __expf(Gs[CHUNK - 1][tid]);
  for (int e = tid; e < CHUNK * HK; e += 256) {
    int t = e >> 6, i = e & 63;
    ksT[i][t] = f2bf(kproj[(size_t)(tok0 + t) * DK + h * HK + i] *
                     __expf(Gs[CHUNK - 1][i] - Gs[t][i]));
  }
  for (int e = tid; e < CHUNK * HV; e += 256) {
    int t = e >> 7, j = e & 127;
    vT[j][t] = f2bf(vproj[(size_t)(tok0 + t) * DV + h * HV + j]);
  }
  __syncthreads();

  // S[64,128] = ksT @ vT^T : M=64,N=128,K=64 -> 4x8 tiles over 8 waves
  const int w = tid >> 5, lane = tid & 31, l = lane & 15, hi = lane >> 4;
  const int mi = w & 3, ni0 = (w >> 2) * 4;
  const __bf16* arow = &ksT[mi * 16 + l][0];
  for (int nt = 0; nt < 4; ++nt) {
    int ni = ni0 + nt;
    const __bf16* brow = &vT[ni * 16 + l][0];
    v8f acc = {};
#pragma unroll
    for (int kt = 0; kt < 2; ++kt) {
      int k0 = kt * 32;
      acc = __builtin_amdgcn_wmma_f32_16x16x32_bf16(false, fragA(arow + k0, hi),
                                                    false, fragB(brow + k0, hi),
                                                    (short)0, acc, false, false);
    }
#pragma unroll
    for (int i = 0; i < 8; ++i) {
      int row = mi * 16 + hi * 8 + i, col = ni * 16 + l;
      Sbuf[gbase * (HK * HV) + (size_t)row * HV + col] = acc[i];
    }
  }
}

// ---------------- sequential diagonal scan: h_{c+1} = d_c (.) h_c + S_c ----------------
__global__ void state_scan_kernel(const float* __restrict__ Sbuf,
                                  const float* __restrict__ dvec,
                                  float* __restrict__ Hbuf) {
  const int bh = blockIdx.x, tid = threadIdx.x;
  __shared__ float dl[HK];
  float hreg[16];
#pragma unroll
  for (int r = 0; r < 16; ++r) hreg[r] = 0.f;
  for (int c = 0; c < NC; ++c) {
    size_t base = (size_t)bh * NC + c;
    if (tid < HK) dl[tid] = dvec[base * HK + tid];
    __syncthreads();
#pragma unroll
    for (int r = 0; r < 16; ++r) {
      int e = tid + 512 * r;
      Hbuf[base * (HK * HV) + e] = hreg[r];
      hreg[r] = dl[e >> 7] * hreg[r] + Sbuf[base * (HK * HV) + e];
    }
    __syncthreads();
  }
}

// ---------------- intra-chunk attention: Attn = mask(Qtil @ Khat^T) -> bf16, Qtil -> bf16 ----------------
__global__ void chunk_attn_kernel(const float* __restrict__ qproj,
                                  const float* __restrict__ kproj,
                                  const float* __restrict__ Gbuf,
                                  __bf16* __restrict__ attnG,
                                  __bf16* __restrict__ qtil) {
  const int c = blockIdx.x, bh = blockIdx.y;
  const int b = bh / HEADS, h = bh % HEADS;
  const int tok0 = b * SEQ + c * CHUNK;
  __shared__ float Gs[CHUNK][HK];
  __shared__ __align__(16) __bf16 qsb[CHUNK][KP];    // A: [t][i]
  __shared__ __align__(16) __bf16 khb[CHUNK][KP];    // B: [s][i]
  const int tid = threadIdx.x;
  const size_t gbase = (size_t)bh * NC + c;

  for (int e = tid; e < CHUNK * HK; e += 256) (&Gs[0][0])[e] = Gbuf[gbase * (CHUNK * HK) + e];
  __syncthreads();
  for (int e = tid; e < CHUNK * HK; e += 256) {
    int t = e >> 6, i = e & 63;
    float g = Gs[t][i];
    __bf16 q = f2bf(qproj[(size_t)(tok0 + t) * DK + h * HK + i] * __expf(g));
    qsb[t][i] = q;
    khb[t][i] = f2bf(kproj[(size_t)(tok0 + t) * DK + h * HK + i] * __expf(-g));
    qtil[gbase * (CHUNK * HK) + (size_t)t * HK + i] = q;
  }
  __syncthreads();

  // Attn[64,64] = qsb @ khb^T : 4x4 tiles over 8 waves (2 each)
  const int w = tid >> 5, lane = tid & 31, l = lane & 15, hi = lane >> 4;
  const int mi = w >> 1, ni0 = (w & 1) * 2;
  const __bf16* arow = &qsb[mi * 16 + l][0];
  for (int nt = 0; nt < 2; ++nt) {
    int ni = ni0 + nt;
    const __bf16* brow = &khb[ni * 16 + l][0];
    v8f acc = {};
#pragma unroll
    for (int kt = 0; kt < 2; ++kt) {
      int k0 = kt * 32;
      acc = __builtin_amdgcn_wmma_f32_16x16x32_bf16(false, fragA(arow + k0, hi),
                                                    false, fragB(brow + k0, hi),
                                                    (short)0, acc, false, false);
    }
#pragma unroll
    for (int i = 0; i < 8; ++i) {
      int row = mi * 16 + hi * 8 + i, col = ni * 16 + l;   // row=t, col=s
      attnG[gbase * (CHUNK * CHUNK) + (size_t)row * CHUNK + col] =
          (col <= row) ? f2bf(acc[i]) : f2bf(0.f);         // causal mask
    }
  }
}

// ---------------- chunk output: O = Attn @ V + Qtil @ h0 (pre-LN, fp32) ----------------
__global__ void chunk_out_kernel(const __bf16* __restrict__ attnG,
                                 const __bf16* __restrict__ qtil,
                                 const float* __restrict__ vproj,
                                 const float* __restrict__ Hbuf,
                                 float* __restrict__ og) {
  const int c = blockIdx.x, bh = blockIdx.y;
  const int b = bh / HEADS, h = bh % HEADS;
  const int tok0 = b * SEQ + c * CHUNK;
  __shared__ __align__(16) __bf16 atb[CHUNK][KP];    // A: [t][s]
  __shared__ __align__(16) __bf16 qsb[CHUNK][KP];    // A: [t][i]
  __shared__ __align__(16) __bf16 vTb[HV][KP];       // B: [j][s]
  __shared__ __align__(16) __bf16 hTb[HV][KP];       // B: [j][i]
  const int tid = threadIdx.x;
  const size_t gbase = (size_t)bh * NC + c;

  for (int e = tid; e < CHUNK * CHUNK; e += 256) {
    int t = e >> 6, s = e & 63;
    atb[t][s] = attnG[gbase * (CHUNK * CHUNK) + e];
    qsb[t][s] = qtil[gbase * (CHUNK * HK) + e];
  }
  for (int e = tid; e < CHUNK * HV; e += 256) {
    int t = e >> 7, j = e & 127;
    vTb[j][t] = f2bf(vproj[(size_t)(tok0 + t) * DV + h * HV + j]);
  }
  for (int e = tid; e < HK * HV; e += 256) {
    int i = e >> 7, j = e & 127;
    hTb[j][i] = f2bf(Hbuf[gbase * (HK * HV) + e]);
  }
  __syncthreads();

  // O[64,128] : 4x8 tiles over 8 waves, 4 accumulating k-steps
  const int w = tid >> 5, lane = tid & 31, l = lane & 15, hi = lane >> 4;
  const int mi = w & 3, ni0 = (w >> 2) * 4;
  const __bf16* arow0 = &atb[mi * 16 + l][0];
  const __bf16* arow1 = &qsb[mi * 16 + l][0];
  for (int nt = 0; nt < 4; ++nt) {
    int ni = ni0 + nt;
    const __bf16* brow0 = &vTb[ni * 16 + l][0];
    const __bf16* brow1 = &hTb[ni * 16 + l][0];
    v8f acc = {};
#pragma unroll
    for (int kt = 0; kt < 2; ++kt) {       // Attn @ V
      int k0 = kt * 32;
      acc = __builtin_amdgcn_wmma_f32_16x16x32_bf16(false, fragA(arow0 + k0, hi),
                                                    false, fragB(brow0 + k0, hi),
                                                    (short)0, acc, false, false);
    }
#pragma unroll
    for (int kt = 0; kt < 2; ++kt) {       // Qtil @ h0
      int k0 = kt * 32;
      acc = __builtin_amdgcn_wmma_f32_16x16x32_bf16(false, fragA(arow1 + k0, hi),
                                                    false, fragB(brow1 + k0, hi),
                                                    (short)0, acc, false, false);
    }
#pragma unroll
    for (int i = 0; i < 8; ++i) {
      int row = mi * 16 + hi * 8 + i, col = ni * 16 + l;
      og[(size_t)(tok0 + row) * DV + h * HV + col] = acc[i];
    }
  }
}

// ---------------- per-(token,head) LayerNorm + SiLU output gate, in place on og ----------------
// one wave per 128-wide row; 8 waves per block
__global__ void ln_gate_kernel(float* __restrict__ og,
                               const float* __restrict__ gproj) {
  const int row = blockIdx.x * 8 + (threadIdx.x >> 5);   // over NTOK*HEADS rows
  const int lane = threadIdx.x & 31;
  const int tok = row >> 2, h = row & 3;                 // HEADS == 4
  const size_t base = (size_t)tok * DV + h * HV;

  float vals[4];
  float s = 0.f;
#pragma unroll
  for (int e = 0; e < 4; ++e) { vals[e] = og[base + lane * 4 + e]; s += vals[e]; }
#pragma unroll
  for (int off = 16; off >= 1; off >>= 1) s += __shfl_xor(s, off, 32);
  float mu = s * (1.f / HV);
  float v2 = 0.f;
#pragma unroll
  for (int e = 0; e < 4; ++e) { float d = vals[e] - mu; v2 += d * d; }
#pragma unroll
  for (int off = 16; off >= 1; off >>= 1) v2 += __shfl_xor(v2, off, 32);
  float rstd = rsqrtf(v2 * (1.f / HV) + 1e-5f);
#pragma unroll
  for (int e = 0; e < 4; ++e) {
    float gg = gproj[base + lane * 4 + e];
    og[base + lane * 4 + e] = (vals[e] - mu) * rstd * (gg / (1.f + __expf(-gg)));
  }
}

// ---------------- host side ----------------
extern "C" void kernel_launch(void* const* d_in, const int* in_sizes, int n_in,
                              void* d_out, int out_size, void* d_ws, size_t ws_size,
                              hipStream_t stream) {
  const float* x    = (const float*)d_in[0];
  const float* Wq   = (const float*)d_in[1];
  const float* Wk   = (const float*)d_in[2];
  const float* Wv   = (const float*)d_in[3];
  const float* Wg   = (const float*)d_in[4];
  const float* Wgk1 = (const float*)d_in[5];
  const float* Wgk2 = (const float*)d_in[6];
  const float* bgk2 = (const float*)d_in[7];
  const float* Wo   = (const float*)d_in[8];

  char* ws = (char*)d_ws;
  size_t off = 0;
  auto alloc = [&](size_t bytes) -> void* {
    off = (off + 255) & ~(size_t)255;
    void* p = ws + off;
    off += bytes;
    return p;
  };

  __bf16* x_bf    = (__bf16*)alloc((size_t)NTOK * Dm * 2);
  __bf16* Wq_bf   = (__bf16*)alloc((size_t)Dm * DK * 2);
  __bf16* Wk_bf   = (__bf16*)alloc((size_t)Dm * DK * 2);
  __bf16* Wv_bf   = (__bf16*)alloc((size_t)Dm * DV * 2);
  __bf16* Wg_bf   = (__bf16*)alloc((size_t)Dm * DV * 2);
  __bf16* Wgk1_bf = (__bf16*)alloc((size_t)Dm * GLR * 2);
  __bf16* Wo_bf   = (__bf16*)alloc((size_t)DV * Dm * 2);
  float*  qb    = (float*)alloc((size_t)NTOK * DK * 4);
  float*  kb    = (float*)alloc((size_t)NTOK * DK * 4);
  float*  vb    = (float*)alloc((size_t)NTOK * DV * 4);
  float*  gb    = (float*)alloc((size_t)NTOK * DV * 4);
  float*  xg1   = (float*)alloc((size_t)NTOK * GLR * 4);
  float*  gkb   = (float*)alloc((size_t)NTOK * DK * 4);
  float*  Gbuf  = (float*)alloc((size_t)BH * NC * CHUNK * HK * 4);
  float*  dvec  = (float*)alloc((size_t)BH * NC * HK * 4);
  float*  Sbuf  = (float*)alloc((size_t)BH * NC * HK * HV * 4);
  float*  Hbuf  = (float*)alloc((size_t)BH * NC * HK * HV * 4);
  __bf16* attnG = (__bf16*)alloc((size_t)BH * NC * CHUNK * CHUNK * 2);
  __bf16* qtil  = (__bf16*)alloc((size_t)BH * NC * CHUNK * HK * 2);
  float*  og    = (float*)alloc((size_t)NTOK * DV * 4);
  __bf16* og_bf = (__bf16*)alloc((size_t)NTOK * DV * 2);
  (void)ws_size; (void)in_sizes; (void)n_in; (void)out_size;

  auto cvt = [&](const float* src, __bf16* dst, size_t n) {
    int grid = (int)((n + 255) / 256);
    if (grid > 8192) grid = 8192;
    cvt_bf16_kernel<<<grid, 256, 0, stream>>>(src, dst, n);
  };
  cvt(x,    x_bf,    (size_t)NTOK * Dm);
  cvt(Wq,   Wq_bf,   (size_t)Dm * DK);
  cvt(Wk,   Wk_bf,   (size_t)Dm * DK);
  cvt(Wv,   Wv_bf,   (size_t)Dm * DV);
  cvt(Wg,   Wg_bf,   (size_t)Dm * DV);
  cvt(Wgk1, Wgk1_bf, (size_t)Dm * GLR);
  cvt(Wo,   Wo_bf,   (size_t)DV * Dm);

  auto gemm = [&](const __bf16* A, const __bf16* B, float* C, int M, float alpha) {
    dim3 grid(M / 16, NTOK / 128);
    gemm_bf16_kernel<<<grid, 256, 0, stream>>>(A, B, C, NTOK, Dm, M, alpha);
  };
  gemm(x_bf, Wq_bf,   qb,  DK,  0.125f);   // HK^-0.5
  gemm(x_bf, Wk_bf,   kb,  DK,  1.f);
  gemm(x_bf, Wv_bf,   vb,  DV,  1.f);
  gemm(x_bf, Wg_bf,   gb,  DV,  1.f);
  gemm(x_bf, Wgk1_bf, xg1, GLR, 1.f);

  {
    size_t tot = (size_t)NTOK * DK;
    gate_lowrank_kernel<<<(int)((tot + 255) / 256), 256, 0, stream>>>(xg1, Wgk2, bgk2, gkb);
  }

  {
    dim3 grid(NC, BH);
    chunk_state_kernel<<<grid, 256, 0, stream>>>(kb, vb, gkb, Sbuf, dvec, Gbuf);
  }
  state_scan_kernel<<<BH, 512, 0, stream>>>(Sbuf, dvec, Hbuf);
  {
    dim3 grid(NC, BH);
    chunk_attn_kernel<<<grid, 256, 0, stream>>>(qb, kb, Gbuf, attnG, qtil);
    chunk_out_kernel<<<grid, 256, 0, stream>>>(attnG, qtil, vb, Hbuf, og);
  }
  ln_gate_kernel<<<(NTOK * HEADS) / 8, 256, 0, stream>>>(og, gb);

  cvt(og, og_bf, (size_t)NTOK * DV);
  {
    dim3 grid(Dm / 16, NTOK / 128);
    gemm_bf16_kernel<<<grid, 256, 0, stream>>>(og_bf, Wo_bf, (float*)d_out,
                                               NTOK, DV, Dm, 1.f);
  }
}
